// FlashPhiAttention_3453153706154
// MI455X (gfx1250) — matrix operand order repaired and checked by reading, not compiled, not run on hardware
//
#include <hip/hip_runtime.h>

// ---------------------------------------------------------------------------
// Problem constants (match reference)
// ---------------------------------------------------------------------------
#define TT   4096          // B*S tokens
#define SS   1024          // sequence length
#define BB   4             // batch
#define HH   32            // heads
#define DD   80            // head dim
#define DP   96            // padded head dim (multiple of 32 for WMMA k-loop)
#define HID  2560
#define N3   7680          // 3*HID
#define NSLOTS 8192        // 2*T
static constexpr float SCALE_F = 0.11180339887498948f; // 80^-0.5

typedef __attribute__((ext_vector_type(16))) _Float16 v16h;
typedef __attribute__((ext_vector_type(8)))  _Float16 v8h;
typedef __attribute__((ext_vector_type(8)))  float    v8f;

union F16x16 { v16h v; v8h h[2]; };

#define WMMA_F16(a,b,c) \
  __builtin_amdgcn_wmma_f32_16x16x32_f16(false,(a),false,(b),(short)0,(c),false,false)

// A fragment (16x32 f16, M x K): lane holds row (lane&15); element e maps to
// k = 16*(e/8) + 8*(lane>>4) + (e%8)  -> two contiguous 16B chunks.
__device__ __forceinline__ v16h load_a_frag(const _Float16* base, int ld, int k0) {
  const int l = threadIdx.x & 31;
  const _Float16* p = base + (size_t)(l & 15) * ld + k0 + 8 * (l >> 4);
  F16x16 f;
  f.h[0] = *(const v8h*)(p);
  f.h[1] = *(const v8h*)(p + 16);
  return f.v;
}

// B fragment (32x16 f16, K x N) loaded from row-major B^T (N x K): lane holds
// column (lane&15); element e maps to k = 16*(e/8) + 8*(lane>>4) + (e%8).
__device__ __forceinline__ v16h load_b_frag(const _Float16* base, int ld, int k0) {
  const int l = threadIdx.x & 31;
  const _Float16* p = base + (size_t)(l & 15) * ld + k0 + 8 * (l >> 4);
  F16x16 f;
  f.h[0] = *(const v8h*)(p);
  f.h[1] = *(const v8h*)(p + 16);
  return f.v;
}

__device__ __forceinline__ void lds_fence() {
  asm volatile("" ::: "memory");
#if __has_builtin(__builtin_amdgcn_s_wait_dscnt)
  __builtin_amdgcn_s_wait_dscnt(0);
#else
  asm volatile("s_wait_dscnt 0x0" ::: "memory");
#endif
  asm volatile("" ::: "memory");
}

// ---------------------------------------------------------------------------
// Elementwise helpers
// ---------------------------------------------------------------------------
__global__ void f32_to_f16_kernel(const float* __restrict__ s,
                                  _Float16* __restrict__ d, int n) {
  int i = blockIdx.x * 256 + threadIdx.x;
  if (i < n) d[i] = (_Float16)s[i];
}

__global__ void copy_f32x4_kernel(const float* __restrict__ s,
                                  float* __restrict__ d, int n4) {
  int i = blockIdx.x * 256 + threadIdx.x;
  if (i < n4) ((float4*)d)[i] = ((const float4*)s)[i];
}

// ---------------------------------------------------------------------------
// WMMA GEMM: C[M,N] = A[M,K] @ W[N,K]^T + bias[N]   (A,W f16; C,bias f32)
// 256 threads = 8 waves stacked in M; each wave owns a 32x64 C tile
// (2 A frags x 4 B frags -> 8 WMMAs per k-step), software-pipelined with
// double-buffered fragments so global loads overlap the WMMAs.
// ---------------------------------------------------------------------------
__global__ void __launch_bounds__(256)
gemm_wmma_kernel(const _Float16* __restrict__ A, const _Float16* __restrict__ W,
                 const float* __restrict__ bias, float* __restrict__ C,
                 int M, int N, int K) {
  const int w    = threadIdx.x >> 5;
  const int lane = threadIdx.x & 31;
  const int n    = lane & 15, hi = lane >> 4;
  const int m0   = blockIdx.y * 256 + w * 32;
  const int n0   = blockIdx.x * 64;

  const _Float16* A0 = A + (size_t)m0 * K;
  const _Float16* A1 = A0 + (size_t)16 * K;
  const _Float16* W0 = W + (size_t)n0 * K;

  const v8f z8 = {0.f,0.f,0.f,0.f,0.f,0.f,0.f,0.f};
  v8f acc0[4], acc1[4];
#pragma unroll
  for (int j = 0; j < 4; ++j) { acc0[j] = z8; acc1[j] = z8; }

  // prologue: fragments for k = 0
  v16h a0 = load_a_frag(A0, K, 0);
  v16h a1 = load_a_frag(A1, K, 0);
  v16h b0 = load_b_frag(W0,                   K, 0);
  v16h b1 = load_b_frag(W0 + (size_t)16 * K,  K, 0);
  v16h b2 = load_b_frag(W0 + (size_t)32 * K,  K, 0);
  v16h b3 = load_b_frag(W0 + (size_t)48 * K,  K, 0);

  for (int k0 = 32; k0 < K; k0 += 32) {
    // issue next-step loads first (independent of current WMMAs)
    v16h a0n = load_a_frag(A0, K, k0);
    v16h a1n = load_a_frag(A1, K, k0);
    v16h b0n = load_b_frag(W0,                  K, k0);
    v16h b1n = load_b_frag(W0 + (size_t)16 * K, K, k0);
    v16h b2n = load_b_frag(W0 + (size_t)32 * K, K, k0);
    v16h b3n = load_b_frag(W0 + (size_t)48 * K, K, k0);

    acc0[0] = WMMA_F16(a0, b0, acc0[0]);
    acc0[1] = WMMA_F16(a0, b1, acc0[1]);
    acc0[2] = WMMA_F16(a0, b2, acc0[2]);
    acc0[3] = WMMA_F16(a0, b3, acc0[3]);
    acc1[0] = WMMA_F16(a1, b0, acc1[0]);
    acc1[1] = WMMA_F16(a1, b1, acc1[1]);
    acc1[2] = WMMA_F16(a1, b2, acc1[2]);
    acc1[3] = WMMA_F16(a1, b3, acc1[3]);

    a0 = a0n; a1 = a1n; b0 = b0n; b1 = b1n; b2 = b2n; b3 = b3n;
  }
  // epilogue: last k-step
  acc0[0] = WMMA_F16(a0, b0, acc0[0]);
  acc0[1] = WMMA_F16(a0, b1, acc0[1]);
  acc0[2] = WMMA_F16(a0, b2, acc0[2]);
  acc0[3] = WMMA_F16(a0, b3, acc0[3]);
  acc1[0] = WMMA_F16(a1, b0, acc1[0]);
  acc1[1] = WMMA_F16(a1, b1, acc1[1]);
  acc1[2] = WMMA_F16(a1, b2, acc1[2]);
  acc1[3] = WMMA_F16(a1, b3, acc1[3]);

#pragma unroll
  for (int j = 0; j < 4; ++j) {
    const int col = n0 + 16 * j + n;
    const float bv = bias[col];
#pragma unroll
    for (int r = 0; r < 8; ++r) {
      const int row = m0 + r + 8 * hi;
      C[(size_t)row * N + col]        = acc0[j][r] + bv;
      C[(size_t)(row + 16) * N + col] = acc1[j][r] + bv;
    }
  }
}

// ---------------------------------------------------------------------------
// RoPE + KV-cache scatter + pack Q/K (padded f16) and V (transposed f16)
// one thread per (t, h, dp) with dp in [0,96)
// ---------------------------------------------------------------------------
__global__ void rope_pack_kernel(const float* __restrict__ qkv,
                                 const float* __restrict__ cosp,
                                 const float* __restrict__ sinp,
                                 const int* __restrict__ slots,
                                 float* __restrict__ kc, float* __restrict__ vc,
                                 _Float16* __restrict__ qh,
                                 _Float16* __restrict__ kh,
                                 _Float16* __restrict__ vt) {
  const int idx = blockIdx.x * 256 + threadIdx.x;
  const int total = TT * HH * DP;
  if (idx >= total) return;
  const int dp = idx % DP;
  const int h  = (idx / DP) % HH;
  const int t  = idx / (DP * HH);
  const int b  = t >> 10;
  const int sq = t & 1023;
  const size_t qkbase = ((size_t)(b * HH + h) * SS + sq) * DP;

  if (dp >= DD) {               // zero padding for WMMA k-dim
    qh[qkbase + dp] = (_Float16)0.f;
    kh[qkbase + dp] = (_Float16)0.f;
    return;
  }
  const int d = dp;
  const float* rowq = qkv + (size_t)t * N3 + h * DD;
  float q = rowq[d];
  float k = rowq[HID + d];
  float v = rowq[2 * HID + d];

  if (d < 32) {                 // partial rope: rotate first 32 dims (16+16)
    if (d < 16) {
      const float c = cosp[t * 16 + d], s = sinp[t * 16 + d];
      const float q2 = rowq[d + 16], k2 = rowq[HID + d + 16];
      q = q * c - q2 * s;
      k = k * c - k2 * s;
    } else {
      const int j = d - 16;
      const float c = cosp[t * 16 + j], s = sinp[t * 16 + j];
      const float q1 = rowq[d - 16], k1 = rowq[HID + d - 16];
      q = q1 * s + q * c;
      k = k1 * s + k * c;
    }
  }

  const int slot = slots[t];
  kc[(size_t)slot * HID + h * DD + d] = k;
  vc[(size_t)slot * HID + h * DD + d] = v;

  qh[qkbase + d] = (_Float16)q;
  kh[qkbase + d] = (_Float16)k;
  vt[((size_t)(b * HH + h) * DD + d) * SS + sq] = (_Float16)v;   // transposed V
}

// ---------------------------------------------------------------------------
// Causal flash attention. 128 threads = 4 independent waves; each wave owns
// one (b, h, 16-row q-tile). K tiles of 32, QK^T via 3 WMMA k-steps (DP=96),
// softmax through per-wave LDS, P@V via 5 WMMAs (D=80 = 5x16 N-chunks).
// Q fragments hoisted out of the loop; V fragments loaded before the softmax
// so their latency hides under the exp/max scalar work.
// ---------------------------------------------------------------------------
__global__ void __launch_bounds__(128)
attn_kernel(const _Float16* __restrict__ qh, const _Float16* __restrict__ kh,
            const _Float16* __restrict__ vt, _Float16* __restrict__ attn) {
  __shared__ float sc[4][16][32];
  __shared__ __align__(32) _Float16 pt[4][16][32];
  __shared__ float red[4][16];

  const int w    = threadIdx.x >> 5;
  const int lane = threadIdx.x & 31;
  const int n    = lane & 15, hi = lane >> 4;

  const int gid = blockIdx.x * 4 + w;            // 8192 wave-tiles total
  const int b   = gid >> 11;
  const int h   = (gid >> 6) & 31;
  const int q0  = (gid & 63) << 4;

  const _Float16* qptr = qh + ((size_t)(b * HH + h) * SS + q0) * DP;
  const _Float16* kptr = kh + (size_t)(b * HH + h) * SS * DP;
  const _Float16* vptr = vt + (size_t)(b * HH + h) * DD * SS;

  float m_i = -3.0e38f, l_i = 0.0f;
  const v8f z8 = {0.f,0.f,0.f,0.f,0.f,0.f,0.f,0.f};
  v8f accO[5];
#pragma unroll
  for (int j = 0; j < 5; ++j) accO[j] = z8;

  // Q fragments are loop-invariant: load once
  v16h aq[3];
#pragma unroll
  for (int ks = 0; ks < 3; ++ks) aq[ks] = load_a_frag(qptr, DP, ks * 32);

  const int kend = q0 + 16;
  for (int kt = 0; kt < kend; kt += 32) {
    // ---- scores: S = Q @ K^T over padded D (3 k-steps), two 16-col tiles
    v8f s0 = z8, s1 = z8;
#pragma unroll
    for (int ks = 0; ks < 3; ++ks) {
      v16h b0 = load_b_frag(kptr + (size_t)kt * DP, DP, ks * 32);
      v16h b1 = load_b_frag(kptr + (size_t)(kt + 16) * DP, DP, ks * 32);
      s0 = WMMA_F16(aq[ks], b0, s0);
      s1 = WMMA_F16(aq[ks], b1, s1);
    }
    // ---- issue V fragment loads now; latency hides under the softmax below
    v16h vb[5];
#pragma unroll
    for (int j = 0; j < 5; ++j)
      vb[j] = load_b_frag(vptr + (size_t)(16 * j) * SS + kt, SS, 0);

    // ---- scale + causal mask, drop into LDS (C layout -> row-major)
#pragma unroll
    for (int r = 0; r < 8; ++r) {
      const int qrow = q0 + r + 8 * hi;
      float x0 = s0[r] * SCALE_F;
      float x1 = s1[r] * SCALE_F;
      if (kt + n > qrow)      x0 = -3.0e38f;
      if (kt + 16 + n > qrow) x1 = -3.0e38f;
      sc[w][r + 8 * hi][n]      = x0;
      sc[w][r + 8 * hi][n + 16] = x1;
    }
    lds_fence();
    // ---- online softmax: lane (n, hi) handles row n, cols [16*hi,16*hi+16)
    float vals[16];
    float lmax = -3.0e38f;
#pragma unroll
    for (int c = 0; c < 16; ++c) {
      vals[c] = sc[w][n][16 * hi + c];
      lmax = fmaxf(lmax, vals[c]);
    }
    lmax = fmaxf(lmax, __shfl_xor(lmax, 16));
    const float m_new = fmaxf(m_i, lmax);
    const float alpha = __expf(m_i - m_new);
    float ps = 0.0f;
#pragma unroll
    for (int c = 0; c < 16; ++c) {
      const float p = __expf(vals[c] - m_new);
      ps += p;
      pt[w][n][16 * hi + c] = (_Float16)p;
    }
    ps += __shfl_xor(ps, 16);
    l_i = l_i * alpha + ps;
    m_i = m_new;
    red[w][n] = alpha;            // both halves write identical value
    lds_fence();
    // ---- rescale accumulators by alpha of their rows
    float al[8];
#pragma unroll
    for (int r = 0; r < 8; ++r) al[r] = red[w][r + 8 * hi];
#pragma unroll
    for (int j = 0; j < 5; ++j)
#pragma unroll
      for (int r = 0; r < 8; ++r) accO[j][r] *= al[r];
    // ---- O += P @ V  (A from LDS, B fragments already in flight)
    v16h ap = load_a_frag(&pt[w][0][0], 32, 0);
#pragma unroll
    for (int j = 0; j < 5; ++j)
      accO[j] = WMMA_F16(ap, vb[j], accO[j]);
  }

  // ---- normalize by row sums and emit f16 attn output [T, HID]
  red[w][n] = l_i;
  lds_fence();
  float li[8];
#pragma unroll
  for (int r = 0; r < 8; ++r) li[r] = red[w][r + 8 * hi];
#pragma unroll
  for (int j = 0; j < 5; ++j) {
#pragma unroll
    for (int r = 0; r < 8; ++r) {
      const int t = b * SS + q0 + r + 8 * hi;
      attn[(size_t)t * HID + h * DD + 16 * j + n] = (_Float16)(accO[j][r] / li[r]);
    }
  }
}

// ---------------------------------------------------------------------------
// Host-side orchestration
// ---------------------------------------------------------------------------
extern "C" void kernel_launch(void* const* d_in, const int* in_sizes, int n_in,
                              void* d_out, int out_size, void* d_ws, size_t ws_size,
                              hipStream_t stream) {
  const float* hs      = (const float*)d_in[0];
  const float* cosp    = (const float*)d_in[1];
  const float* sinp    = (const float*)d_in[2];
  const int*   slots   = (const int*)  d_in[3];
  const float* kc_in   = (const float*)d_in[4];
  const float* vc_in   = (const float*)d_in[5];
  const float* qkv_w   = (const float*)d_in[6];
  const float* qkv_b   = (const float*)d_in[7];
  const float* dense_w = (const float*)d_in[8];
  const float* dense_b = (const float*)d_in[9];

  float* out = (float*)d_out;
  float* kc  = out + (size_t)TT * HID;            // 10,485,760
  float* vc  = kc + (size_t)NSLOTS * HID;         // +20,971,520

  char* ws = (char*)d_ws;
  _Float16* hs_h   = (_Float16*)(ws);                 // T*HID f16      (20.97 MB)
  _Float16* wqkv_h = (_Float16*)(ws + 20971520ull);   // N3*HID f16     (39.32 MB)
  _Float16* wden_h = (_Float16*)(ws + 60293120ull);   // HID*HID f16    (13.11 MB)
  _Float16* attn_h = (_Float16*)(ws + 73400320ull);   // T*HID f16      (20.97 MB)
  _Float16* q_h    = (_Float16*)(ws + 94371840ull);   // B*H*S*DP f16   (25.17 MB)
  _Float16* k_h    = (_Float16*)(ws + 119537664ull);  // B*H*S*DP f16   (25.17 MB)
  _Float16* v_t    = (_Float16*)(ws + 144703488ull);  // B*H*DD*S f16   (20.97 MB)
  float*    qkvf   = (float*)   (ws + 165675008ull);  // T*N3 f32       (125.8 MB)

  // seed output caches from input caches (kc = cache.at[slots].set(k))
  copy_f32x4_kernel<<<20480, 256, 0, stream>>>(kc_in, kc, 5242880);
  copy_f32x4_kernel<<<20480, 256, 0, stream>>>(vc_in, vc, 5242880);

  // f32 -> f16 conversions for WMMA operands
  f32_to_f16_kernel<<<40960, 256, 0, stream>>>(hs, hs_h, TT * HID);
  f32_to_f16_kernel<<<76800, 256, 0, stream>>>(qkv_w, wqkv_h, N3 * HID);
  f32_to_f16_kernel<<<25600, 256, 0, stream>>>(dense_w, wden_h, HID * HID);

  // qkv = hs @ qkv_w^T + qkv_b   (M=4096, N=7680, K=2560)
  gemm_wmma_kernel<<<dim3(120, 16), 256, 0, stream>>>(
      hs_h, wqkv_h, qkv_b, qkvf, TT, N3, HID);

  // rope + cache scatter + pack for attention
  rope_pack_kernel<<<49152, 256, 0, stream>>>(
      qkvf, cosp, sinp, slots, kc, vc, q_h, k_h, v_t);

  // causal flash attention
  attn_kernel<<<2048, 128, 0, stream>>>(q_h, k_h, v_t, attn_h);

  // out = attn @ dense_w^T + dense_b   (M=4096, N=2560, K=2560)
  gemm_wmma_kernel<<<dim3(40, 16), 256, 0, stream>>>(
      attn_h, wden_h, dense_b, out, TT, HID, HID);
}
